// SyncProjectionHead_31722628448768
// MI455X (gfx1250) — matrix operand order repaired and checked by reading, not compiled, not run on hardware
//
#include <hip/hip_runtime.h>

typedef __attribute__((ext_vector_type(16))) _Float16     v16h;
typedef __attribute__((ext_vector_type(8)))  float        v8f;
typedef __attribute__((ext_vector_type(4)))  unsigned int u32x4;
typedef __attribute__((ext_vector_type(8)))  int          i32x8;
typedef __attribute__((ext_vector_type(4)))  int          i32x4;

#define B_    512
#define T_    512
#define D_    128
#define DOUT_ 32
#define C_    4
#define GAMMA_ 0.9f
#define ALPHA_ 0.1f

#if __has_builtin(__builtin_amdgcn_tensor_load_to_lds) && \
    __has_builtin(__builtin_amdgcn_s_wait_tensorcnt)
#define USE_TDM 1
#else
#define USE_TDM 0
#endif

#define ROW_DW 132               // 128 data dwords + 4 pad dwords (TDM pad feature)
#define BUF_DW (16 * ROW_DW + 4) // one 16-row z tile, incl. trailing pad
#define XS_    36                // x-transpose row stride (dwords): b128-aligned, conflict-free

__global__ __launch_bounds__(32)
void sync_proj_head_scan(const float* __restrict__ z,
                         const float* __restrict__ proj,
                         const float* __restrict__ lnw,
                         const float* __restrict__ lnb,
                         const float* __restrict__ clsw,
                         const float* __restrict__ clsb,
                         float* __restrict__ outx,
                         float* __restrict__ outl) {
  const int lane = threadIdx.x;     // 0..31 (wave32)
  const int half = lane >> 4;       // which K-half this lane owns
  const int n0   = lane & 15;       // row (A) / column (B,C,D) id
  const int m0   = blockIdx.x * 16; // batch-row tile base

  // proj pre-packed in f16 WMMA B-fragment layout: [4 K-chunks][2 N-tiles][lane][16 elems]
  __shared__ __align__(32) _Float16 sB[8 * 32 * 16];
  __shared__ __align__(16) float sX[16 * XS_];  // per-step x tile, transposed access
  __shared__ __align__(16) float sWC[4 * 32];   // wc[k][n] = ln_w[n] * cls_w[k][n]
#if USE_TDM
  __shared__ __align__(16) float sZ[2][BUF_DW]; // double-buffered z tiles (TDM dest)
#endif

#pragma unroll
  for (int c = 0; c < 4; ++c)
#pragma unroll
    for (int tile = 0; tile < 2; ++tile)
#pragma unroll
      for (int e = 0; e < 16; ++e) {
        const int kb = 32 * c + 8 * half + (e < 8 ? e : e + 8); // K index for B fragment
        const int n  = n0 + 16 * tile;              // output column
        sB[((c * 2 + tile) * 32 + lane) * 16 + e] = (_Float16)proj[kb * DOUT_ + n];
      }
#pragma unroll
  for (int k = 0; k < 4; ++k)
    sWC[k * 32 + lane] = lnw[lane] * clsw[k * DOUT_ + lane];
  __syncthreads();

  // Folded per-class constants: p_k = rstd*(S_k - mu*Wk) + Bk
  float Wk[4], Bk[4];
#pragma unroll
  for (int k = 0; k < 4; ++k) {
    float wk = 0.0f, bk = 0.0f;
    for (int n = 0; n < DOUT_; ++n) {
      wk += lnw[n] * clsw[k * DOUT_ + n];
      bk += lnb[n] * clsw[k * DOUT_ + n];
    }
    Wk[k] = wk;
    Bk[k] = bk + clsb[k];
  }

#if USE_TDM
  const unsigned lds0 = (unsigned)(unsigned long long)(uintptr_t)&sZ[0][0];
  const unsigned lds1 = (unsigned)(unsigned long long)(uintptr_t)&sZ[1][0];
  const unsigned long long gbase =
      (unsigned long long)(uintptr_t)(z + (size_t)m0 * T_ * D_);

  // D# group1: data_size=4B(code 2); pad 4 dwords (code 3) every 128 dwords (code 6);
  // tensor 128x16, tile 128x16, dim0 stride = T*D.
  i32x8 g1;
  g1[0] = (int)((2u << 16) | (1u << 20) | (6u << 22) | (3u << 25));
  g1[1] = (int)(128u << 16);   // tensor_dim0[15:0] -> bits 63:48
  g1[2] = (int)(16u << 16);    // tensor_dim1[15:0] -> bits 95:80
  g1[3] = (int)(128u << 16);   // tile_dim0        -> bits 127:112
  g1[4] = 16;                  // tile_dim1        -> bits 143:128
  g1[5] = T_ * D_;             // tensor_dim0_stride[31:0]
  g1[6] = 0;
  g1[7] = 0;
  i32x4 gz4;
  gz4[0] = 0; gz4[1] = 0; gz4[2] = 0; gz4[3] = 0;
  i32x8 gz8;
#pragma unroll
  for (int q = 0; q < 8; ++q) gz8[q] = 0;

  auto tdm_issue = [&](int t, unsigned lds) {
    const unsigned long long ga = gbase + (unsigned long long)t * (D_ * 4);
    u32x4 g0;
    g0[0] = 1u;                                                 // count=1 (valid user D#)
    g0[1] = lds;                                                // lds_addr
    g0[2] = (unsigned)ga;                                       // global_addr[31:0]
    g0[3] = (unsigned)((ga >> 32) & 0x1FFFFFFu) | (2u << 30);   // addr[56:32] | type=2
    __builtin_amdgcn_tensor_load_to_lds(g0, g1, gz4, gz4, gz8, 0);
  };
  tdm_issue(0, lds0);
#else
  const float* zrow = z + (size_t)(m0 + n0) * T_ * D_;
#endif

  // Welford state in A-fragment order: elem i -> chunk c=i>>4, e=i&15,
  // K = 32c + 8*half + (e<8 ? e : e+8)
  float m[64], m2[64];
#pragma unroll
  for (int i = 0; i < 64; ++i) { m[i] = 0.0f; m2[i] = 0.0f; }
  float h[2][8];
#pragma unroll
  for (int tl = 0; tl < 2; ++tl)
#pragma unroll
    for (int i = 0; i < 8; ++i) h[tl][i] = 0.0f;

  float s = 0.0f, tau = 0.0f;

  for (int t = 0; t < T_; ++t) {
    tau += 1.0f;
    const float inv_tau = __builtin_amdgcn_rcpf(tau);
    const float inv_den = __builtin_amdgcn_rcpf(fmaxf(tau - 1.0f, 1.0f));

#if USE_TDM
    // Kick off t+1's DMA, then wait until only it is outstanding (TENSORcnt in-order).
    if (t + 1 < T_) {
      tdm_issue(t + 1, (t & 1) ? lds0 : lds1);
      __builtin_amdgcn_s_wait_tensorcnt(1);
    } else {
      __builtin_amdgcn_s_wait_tensorcnt(0);
    }
    const float* zp = &sZ[t & 1][n0 * ROW_DW];   // this lane's row (padded stride)
#else
    const float* zp = zrow + (size_t)t * D_;
    if (t + 1 < T_) __builtin_prefetch(zp + D_, 0, 0);
#endif

    // ---- Welford update; stash delta2 = (z - m_new) directly into A fragments ----
    float vsum = 0.0f;
    v16h a[4];
#pragma unroll
    for (int c = 0; c < 4; ++c) {
      float zv[16];
#pragma unroll
      for (int r = 0; r < 2; ++r) {
        const float* rb = zp + 32 * c + 16 * r + 8 * half;
        float4 q0 = *(const float4*)(rb);
        float4 q1 = *(const float4*)(rb + 4);
        zv[r * 8 + 0] = q0.x; zv[r * 8 + 1] = q0.y; zv[r * 8 + 2] = q0.z; zv[r * 8 + 3] = q0.w;
        zv[r * 8 + 4] = q1.x; zv[r * 8 + 5] = q1.y; zv[r * 8 + 6] = q1.z; zv[r * 8 + 7] = q1.w;
      }
#pragma unroll
      for (int e = 0; e < 16; ++e) {
        const int i = c * 16 + e;
        const float delta = zv[e] - m[i];
        const float mn    = m[i] + delta * inv_tau;
        const float d2    = zv[e] - mn;
        m[i]  = mn;
        m2[i] = m2[i] + delta * d2;
        vsum += m2[i];
        a[c][e] = (_Float16)d2;
      }
    }

    // v_bar: lane ^ 16 holds the other K-half of the same row
    const float vbar = (vsum + __shfl_xor(vsum, 16, 32)) * inv_den * (1.0f / 128.0f);

    // scale fragments: z_tilde = delta2 * rsqrt((1-a)*v + a*v_bar + eps)
#pragma unroll
    for (int c = 0; c < 4; ++c)
#pragma unroll
      for (int e = 0; e < 16; ++e) {
        const int i = c * 16 + e;
        const float v  = m2[i] * inv_den;
        const float sc = __builtin_amdgcn_rsqf((1.0f - ALPHA_) * v + ALPHA_ * vbar + 1e-5f);
        a[c][e] = (_Float16)((float)a[c][e] * sc);
      }

    // ---- r = z_tilde @ proj : 2 N-tiles x 4 chained K-chunk WMMAs ----
    v8f acc[2];
#pragma unroll
    for (int tile = 0; tile < 2; ++tile) {
      v8f accv = {};
#pragma unroll
      for (int c = 0; c < 4; ++c) {
        v16h bf = *(const v16h*)&sB[((c * 2 + tile) * 32 + lane) * 16];
        accv = __builtin_amdgcn_wmma_f32_16x16x32_f16(
            false, a[c], false, bf, (short)0, accv, false, false);
      }
      acc[tile] = accv;
    }

    // ---- EMA + log-features; scatter x tile into LDS for row-major access ----
    s = GAMMA_ * s + 1.0f;
    const float inv_s = __builtin_amdgcn_rcpf(s);

#pragma unroll
    for (int i = 0; i < 8; ++i) {
      const float r0 = acc[0][i];
      const float r1 = acc[1][i];
      h[0][i] = GAMMA_ * h[0][i] + r0 * r0;
      h[1][i] = GAMMA_ * h[1][i] + r1 * r1;
      const float x0 = __builtin_amdgcn_logf(h[0][i] * inv_s + 1e-6f);
      const float x1 = __builtin_amdgcn_logf(h[1][i] * inv_s + 1e-6f);
      const int rowD = i + 8 * half;             // C/D layout row
      sX[rowD * XS_ + n0]      = x0;
      sX[rowD * XS_ + n0 + 16] = x1;
    }

    // ---- per-row LN + classifier, fully in-lane (lanes 16..31 duplicate rows) ----
    float4 xq[8];
    const float* xrowp = &sX[n0 * XS_];
#pragma unroll
    for (int q = 0; q < 8; ++q) xq[q] = *(const float4*)(xrowp + 4 * q);

    float sum = 0.0f, sums = 0.0f;
    float S0 = 0.0f, S1 = 0.0f, S2 = 0.0f, S3 = 0.0f;
#pragma unroll
    for (int q = 0; q < 8; ++q) {
      const float4 xv = xq[q];
      const float4 c0 = *(const float4*)&sWC[0 * 32 + 4 * q];
      const float4 c1 = *(const float4*)&sWC[1 * 32 + 4 * q];
      const float4 c2 = *(const float4*)&sWC[2 * 32 + 4 * q];
      const float4 c3 = *(const float4*)&sWC[3 * 32 + 4 * q];
      sum  += xv.x + xv.y + xv.z + xv.w;
      sums += xv.x * xv.x + xv.y * xv.y + xv.z * xv.z + xv.w * xv.w;
      S0   += xv.x * c0.x + xv.y * c0.y + xv.z * c0.z + xv.w * c0.w;
      S1   += xv.x * c1.x + xv.y * c1.y + xv.z * c1.z + xv.w * c1.w;
      S2   += xv.x * c2.x + xv.y * c2.y + xv.z * c2.z + xv.w * c2.w;
      S3   += xv.x * c3.x + xv.y * c3.y + xv.z * c3.z + xv.w * c3.w;
    }
    const float mu   = sum * (1.0f / 32.0f);
    const float var  = sums * (1.0f / 32.0f) - mu * mu;
    const float rstd = __builtin_amdgcn_rsqf(var + 1e-5f);

    if (half == 0) {
      const size_t rb = (size_t)(m0 + n0) * T_ + t;
      float4 pv;
      pv.x = rstd * (S0 - mu * Wk[0]) + Bk[0];
      pv.y = rstd * (S1 - mu * Wk[1]) + Bk[1];
      pv.z = rstd * (S2 - mu * Wk[2]) + Bk[2];
      pv.w = rstd * (S3 - mu * Wk[3]) + Bk[3];
      *(float4*)&outl[rb * C_] = pv;

      float* xdst = &outx[rb * DOUT_];           // contiguous 128B row
#pragma unroll
      for (int q = 0; q < 8; ++q) *(float4*)(xdst + 4 * q) = xq[q];
    }
  }
}

extern "C" void kernel_launch(void* const* d_in, const int* in_sizes, int n_in,
                              void* d_out, int out_size, void* d_ws, size_t ws_size,
                              hipStream_t stream) {
  const float* z    = (const float*)d_in[0];
  const float* proj = (const float*)d_in[1];
  const float* lnw  = (const float*)d_in[2];
  const float* lnb  = (const float*)d_in[3];
  const float* clsw = (const float*)d_in[4];
  const float* clsb = (const float*)d_in[5];
  float* outx = (float*)d_out;
  float* outl = outx + (size_t)B_ * T_ * DOUT_;

  sync_proj_head_scan<<<dim3(B_ / 16), dim3(32), 0, stream>>>(
      z, proj, lnw, lnb, clsw, clsb, outx, outl);
}
